// PurePythonLSTM_6330781794640
// MI455X (gfx1250) — matrix operand order repaired and checked by reading, not compiled, not run on hardware
//
#include <hip/hip_runtime.h>

// ---------------------------------------------------------------------------
// Types for CDNA5 WMMA
// ---------------------------------------------------------------------------
typedef __bf16 v16bf __attribute__((ext_vector_type(16)));
typedef float  v8f   __attribute__((ext_vector_type(8)));
typedef unsigned int u32x4 __attribute__((ext_vector_type(4)));
typedef int v4i __attribute__((vector_size(4 * sizeof(int))));

union Frag16 {
    v16bf b;
    u32x4 q[2];   // 32 bytes = 8 VGPRs
};

#define AS1 __attribute__((address_space(1)))
#define AS3 __attribute__((address_space(3)))

// ---------------------------------------------------------------------------
// gfx1250 async global->LDS copy (ASYNCcnt-tracked); fallback = sync copy.
// ---------------------------------------------------------------------------
#if __has_builtin(__builtin_amdgcn_global_load_async_to_lds_b128)
#define HAVE_ASYNC_LDS 1
#else
#define HAVE_ASYNC_LDS 0
#endif

__device__ __forceinline__ void async_copy16(const unsigned short* g, unsigned short* l) {
#if HAVE_ASYNC_LDS
    __builtin_amdgcn_global_load_async_to_lds_b128(
        (AS1 v4i*)g, (AS3 v4i*)l, 0, 0);
#else
    *(u32x4*)l = *(const u32x4*)g;
#endif
}

__device__ __forceinline__ void wait_async0() {
#if HAVE_ASYNC_LDS
#if __has_builtin(__builtin_amdgcn_s_wait_asynccnt)
    __builtin_amdgcn_s_wait_asynccnt(0);
#else
    asm volatile("s_wait_asynccnt 0x0" ::: "memory");
#endif
#endif
}

// fp32 -> bf16 round-to-nearest-even (bit trick)
__device__ __forceinline__ unsigned short f2bf(float f) {
    unsigned int u = __float_as_uint(f);
    unsigned int r = u + 0x7FFFu + ((u >> 16) & 1u);
    return (unsigned short)(r >> 16);
}
__device__ __forceinline__ float bf2f(unsigned short s) {
    return __uint_as_float(((unsigned int)s) << 16);
}

__device__ __forceinline__ float sigm_f(float x) {
    return 1.0f / (1.0f + __expf(-x));
}
__device__ __forceinline__ float tanh_f(float x) {
    x = fminf(fmaxf(x, -15.0f), 15.0f);
    float e = __expf(-2.0f * x);
    return (1.0f - e) / (1.0f + e);
}

// ---------------------------------------------------------------------------
// Prep kernels
// ---------------------------------------------------------------------------

// emb fp32 [64M] -> bf16, 8 elems/thread, vectorized in and out.
__global__ __launch_bounds__(256) void cvt_emb_kernel(
    const float* __restrict__ in, unsigned short* __restrict__ out)
{
    size_t i = ((size_t)blockIdx.x * 256 + threadIdx.x) * 8;
    float4 a = *(const float4*)(in + i);
    float4 b = *(const float4*)(in + i + 4);
    union { unsigned short s[8]; u32x4 q; } t;
    t.s[0] = f2bf(a.x); t.s[1] = f2bf(a.y); t.s[2] = f2bf(a.z); t.s[3] = f2bf(a.w);
    t.s[4] = f2bf(b.x); t.s[5] = f2bf(b.y); t.s[6] = f2bf(b.z); t.s[7] = f2bf(b.w);
    *(u32x4*)(out + i) = t.q;
}

// Concatenate 4 [1024x1024] fp32 matrices into one [4096][1024] bf16 matrix.
__global__ __launch_bounds__(256) void cvt4_bf16_kernel(
    const float* __restrict__ w0, const float* __restrict__ w1,
    const float* __restrict__ w2, const float* __restrict__ w3,
    unsigned short* __restrict__ out)
{
    size_t i = ((size_t)blockIdx.x * 256 + threadIdx.x) * 8;   // over 4*2^20
    int g = (int)(i >> 20);
    size_t r = i & 0xFFFFF;
    const float* w = (g == 0) ? w0 : (g == 1) ? w1 : (g == 2) ? w2 : w3;
    float4 a = *(const float4*)(w + r);
    float4 b = *(const float4*)(w + r + 4);
    union { unsigned short s[8]; u32x4 q; } t;
    t.s[0] = f2bf(a.x); t.s[1] = f2bf(a.y); t.s[2] = f2bf(a.z); t.s[3] = f2bf(a.w);
    t.s[4] = f2bf(b.x); t.s[5] = f2bf(b.y); t.s[6] = f2bf(b.z); t.s[7] = f2bf(b.w);
    *(u32x4*)(out + i) = t.q;
}

// Concatenate 4 bias vectors [1024] into bx[4096] (fp32).
__global__ __launch_bounds__(256) void concat_bias_kernel(
    const float* __restrict__ b0, const float* __restrict__ b1,
    const float* __restrict__ b2, const float* __restrict__ b3,
    float* __restrict__ out)
{
    int i = blockIdx.x * 256 + threadIdx.x;  // 0..4095
    int g = i >> 10;
    int r = i & 1023;
    const float* b = (g == 0) ? b0 : (g == 1) ? b1 : (g == 2) ? b2 : b3;
    out[i] = b[r];
}

// Zero h (both bf16 buffers) and c (fp32). 65536 elements each.
__global__ __launch_bounds__(256) void zero_state_kernel(
    unsigned short* __restrict__ h0, unsigned short* __restrict__ h1,
    float* __restrict__ c)
{
    int i = blockIdx.x * 256 + threadIdx.x;
    h0[i] = 0;
    h1[i] = 0;
    c[i]  = 0.0f;
}

// ---------------------------------------------------------------------------
// Phase 1: x_proj[s][b][n] = emb[b][s][:] @ Wx[n][:] + bx[n]   (bf16 out)
// M = 65536, N = 4096, K = 1024.  Block: 256 thr = 8 waves (4x2),
// BM=128, BN=128, BK=32.  Double-buffered LDS fed by async global->LDS b128.
// ---------------------------------------------------------------------------
__global__ __launch_bounds__(256) void xproj_gemm_kernel(
    const unsigned short* __restrict__ embb, // [65536][1024] bf16
    const unsigned short* __restrict__ Wxb,  // [4096][1024] bf16
    const float* __restrict__ bx,            // [4096]
    unsigned short* __restrict__ xproj)      // [1024][64][4096] bf16
{
    __shared__ __align__(16) unsigned short As[2][128][40];  // BM x BK (+pad)
    __shared__ __align__(16) unsigned short Bs[2][128][40];  // BN x BK (+pad)

    const int tid   = threadIdx.x;
    const int lane  = tid & 31;
    const int wave  = tid >> 5;
    const int wm    = wave >> 1;           // 0..3  -> 32 rows / wave
    const int wn    = wave & 1;            // 0..1  -> 64 cols / wave
    const int hi    = lane >> 4;
    const int ll    = lane & 15;

    const int m_blk = blockIdx.y * 128;
    const int n_blk = blockIdx.x * 128;

    // staging: each thread copies 2x16B of one row (16 contiguous K elems)
    const int srow  = tid >> 1;            // 0..127
    const int shalf = (tid & 1) * 16;      // 0 or 16

    const unsigned short* gA0 = embb + (size_t)(m_blk + srow) * 1024 + shalf;
    const unsigned short* gB0 = Wxb  + (size_t)(n_blk + srow) * 1024 + shalf;

    v8f acc[2][4] = {};                    // [m-tile][n-tile]

    // prologue: stage k=0 into buffer 0
    async_copy16(gA0 + 0, &As[0][srow][shalf]);
    async_copy16(gA0 + 8, &As[0][srow][shalf + 8]);
    async_copy16(gB0 + 0, &Bs[0][srow][shalf]);
    async_copy16(gB0 + 8, &Bs[0][srow][shalf + 8]);
    wait_async0();
    __syncthreads();

    int buf = 0;
    for (int k0 = 0; k0 < 1024; k0 += 32) {
        const int nbuf = buf ^ 1;

        // prefetch next K tile into the other buffer (async, overlaps WMMA)
        if (k0 + 32 < 1024) {
            const unsigned short* ga = gA0 + k0 + 32;
            const unsigned short* gb = gB0 + k0 + 32;
            async_copy16(ga + 0, &As[nbuf][srow][shalf]);
            async_copy16(ga + 8, &As[nbuf][srow][shalf + 8]);
            async_copy16(gb + 0, &Bs[nbuf][srow][shalf]);
            async_copy16(gb + 8, &Bs[nbuf][srow][shalf + 8]);
        }

        // ---- load fragments from current buffer ----
        Frag16 A[2], B[4];
        #pragma unroll
        for (int mt = 0; mt < 2; ++mt) {
            const unsigned short* p = &As[buf][wm * 32 + mt * 16 + ll][hi * 8];
            A[mt].q[0] = *(const u32x4*)(p);
            A[mt].q[1] = *(const u32x4*)(p + 16);
        }
        #pragma unroll
        for (int nt = 0; nt < 4; ++nt) {
            const unsigned short* p = &Bs[buf][wn * 64 + nt * 16 + ll][hi * 8];
            B[nt].q[0] = *(const u32x4*)(p);
            B[nt].q[1] = *(const u32x4*)(p + 16);
        }

        // ---- 8 WMMA per wave per K-chunk ----
        #pragma unroll
        for (int mt = 0; mt < 2; ++mt)
            #pragma unroll
            for (int nt = 0; nt < 4; ++nt)
                acc[mt][nt] = __builtin_amdgcn_wmma_f32_16x16x32_bf16(
                    false, A[mt].b, false, B[nt].b,
                    (short)0, acc[mt][nt], false, false);

        // my async writes into nbuf are done; barrier makes them visible and
        // guarantees every wave has consumed `buf` into registers.
        wait_async0();
        __syncthreads();
        buf = nbuf;
    }

    // ---- epilogue: add bias, convert to bf16, store to [s][b][n] ----
    #pragma unroll
    for (int nt = 0; nt < 4; ++nt) {
        const int n = n_blk + wn * 64 + nt * 16 + ll;
        const float bias = bx[n];
        #pragma unroll
        for (int mt = 0; mt < 2; ++mt) {
            v8f a = acc[mt][nt];
            #pragma unroll
            for (int v = 0; v < 8; ++v) {
                int gm = m_blk + wm * 32 + mt * 16 + v + hi * 8; // = b*1024 + s
                int b  = gm >> 10;
                int s  = gm & 1023;
                xproj[((size_t)s * 64 + b) * 4096 + n] = f2bf(a[v] + bias);
            }
        }
    }
}

// ---------------------------------------------------------------------------
// Phase 2: one LSTM time step.
// gates[64][4096] = xproj_t + h_in @ Wh^T ; elementwise update of c, h.
// Grid: 8 blocks x 256 threads = 64 waves; wave w owns d-columns [16w,16w+16).
// Each wave computes all 4 gates for its slice: 16 accumulators, 16 WMMA / K32.
// Wh (8 MB bf16) stays resident in the 192 MB L2 across all 1024 steps.
// ---------------------------------------------------------------------------
__global__ __launch_bounds__(256) void lstm_step_kernel(
    const unsigned short* __restrict__ xproj_t, // [64][4096] bf16
    const unsigned short* __restrict__ Whb,     // [4096][1024] bf16
    const unsigned short* __restrict__ h_in,    // [64][1024] bf16
    unsigned short* __restrict__ h_out,         // [64][1024] bf16
    float* __restrict__ cf,                     // [64][1024] fp32
    float* __restrict__ hf_out,                 // [64][1024] fp32 (d_out)
    int write_f32)
{
    const int tid  = threadIdx.x;
    const int lane = tid & 31;
    const int w    = blockIdx.x * 8 + (tid >> 5); // 0..63
    const int d0   = w * 16;
    const int hi   = lane >> 4;
    const int ll   = lane & 15;

    v8f acc[4][4];  // [gate][m-tile]

    // init accumulators from x_proj[t] (bf16 -> f32)
    #pragma unroll
    for (int g = 0; g < 4; ++g) {
        const int n = g * 1024 + d0 + ll;
        #pragma unroll
        for (int mt = 0; mt < 4; ++mt) {
            #pragma unroll
            for (int v = 0; v < 8; ++v) {
                int m = mt * 16 + v + hi * 8;
                acc[g][mt][v] = bf2f(xproj_t[(size_t)m * 4096 + n]);
            }
        }
    }

    // K loop over h: 16 WMMA per 32-wide K chunk, fragments straight from L2.
    for (int k0 = 0; k0 < 1024; k0 += 32) {
        Frag16 A[4], B[4];
        #pragma unroll
        for (int mt = 0; mt < 4; ++mt) {
            const unsigned short* p = h_in + (size_t)(mt * 16 + ll) * 1024 + k0 + hi * 8;
            A[mt].q[0] = *(const u32x4*)(p);
            A[mt].q[1] = *(const u32x4*)(p + 16);
        }
        #pragma unroll
        for (int g = 0; g < 4; ++g) {
            const unsigned short* p = Whb + (size_t)(g * 1024 + d0 + ll) * 1024 + k0 + hi * 8;
            B[g].q[0] = *(const u32x4*)(p);
            B[g].q[1] = *(const u32x4*)(p + 16);
        }
        #pragma unroll
        for (int g = 0; g < 4; ++g)
            #pragma unroll
            for (int mt = 0; mt < 4; ++mt)
                acc[g][mt] = __builtin_amdgcn_wmma_f32_16x16x32_bf16(
                    false, A[mt].b, false, B[g].b,
                    (short)0, acc[g][mt], false, false);
    }

    // fused LSTM cell update for this wave's d-slice
    const int d = d0 + ll;
    #pragma unroll
    for (int mt = 0; mt < 4; ++mt) {
        #pragma unroll
        for (int v = 0; v < 8; ++v) {
            int m = mt * 16 + v + hi * 8;
            size_t idx = (size_t)m * 1024 + d;
            float i_g = sigm_f(acc[0][mt][v]);
            float f_g = sigm_f(acc[1][mt][v]);
            float g_g = tanh_f(acc[2][mt][v]);
            float o_g = sigm_f(acc[3][mt][v]);
            float c_n = f_g * cf[idx] + i_g * g_g;
            cf[idx] = c_n;
            float h_n = o_g * tanh_f(c_n);
            h_out[idx] = f2bf(h_n);
            if (write_f32) hf_out[idx] = h_n;
        }
    }
}

// ---------------------------------------------------------------------------
// Host launcher
// ---------------------------------------------------------------------------
extern "C" void kernel_launch(void* const* d_in, const int* in_sizes, int n_in,
                              void* d_out, int out_size, void* d_ws, size_t ws_size,
                              hipStream_t stream) {
    (void)in_sizes; (void)n_in; (void)out_size; (void)ws_size;

    const float* emb  = (const float*)d_in[0];
    const float* W_ii = (const float*)d_in[1];
    const float* b_ii = (const float*)d_in[2];
    const float* W_if = (const float*)d_in[3];
    const float* b_if = (const float*)d_in[4];
    const float* W_ig = (const float*)d_in[5];
    const float* b_ig = (const float*)d_in[6];
    const float* W_io = (const float*)d_in[7];
    const float* b_io = (const float*)d_in[8];
    const float* W_hi = (const float*)d_in[9];
    const float* W_hf = (const float*)d_in[10];
    const float* W_hg = (const float*)d_in[11];
    const float* W_ho = (const float*)d_in[12];

    char* ws = (char*)d_ws;
    size_t off = 0;
    unsigned short* Wxb  = (unsigned short*)(ws + off); off += (size_t)4096 * 1024 * 2;  // 8 MiB
    unsigned short* Whb  = (unsigned short*)(ws + off); off += (size_t)4096 * 1024 * 2;  // 8 MiB
    float*          bx   = (float*)         (ws + off); off += (size_t)4096 * 4;         // 16 KiB
    unsigned short* hb0  = (unsigned short*)(ws + off); off += (size_t)64 * 1024 * 2;    // 128 KiB
    unsigned short* hb1  = (unsigned short*)(ws + off); off += (size_t)64 * 1024 * 2;    // 128 KiB
    float*          cf   = (float*)         (ws + off); off += (size_t)64 * 1024 * 4;    // 256 KiB
    unsigned short* embb = (unsigned short*)(ws + off); off += (size_t)65536 * 1024 * 2; // 128 MiB
    unsigned short* xproj = (unsigned short*)(ws + off);  // 512 MiB, [S][B][4D] bf16

    // prep: emb & weights -> bf16, bias concat, zero state
    cvt_emb_kernel<<<32768, 256, 0, stream>>>(emb, embb);          // 64M elems, 8/thr
    cvt4_bf16_kernel<<<2048, 256, 0, stream>>>(W_ii, W_if, W_ig, W_io, Wxb);
    cvt4_bf16_kernel<<<2048, 256, 0, stream>>>(W_hi, W_hf, W_hg, W_ho, Whb);
    concat_bias_kernel<<<16, 256, 0, stream>>>(b_ii, b_if, b_ig, b_io, bx);
    zero_state_kernel<<<256, 256, 0, stream>>>(hb0, hb1, cf);

    // phase 1: big WMMA GEMM for all input projections
    dim3 grid1(4096 / 128, 65536 / 128);   // (N tiles, M tiles) = (32, 512)
    xproj_gemm_kernel<<<grid1, 256, 0, stream>>>(embb, Wxb, bx, xproj);

    // phase 2: 1024 serial steps, double-buffered h
    unsigned short* hin  = hb0;
    unsigned short* hout = hb1;
    for (int t = 0; t < 1024; ++t) {
        lstm_step_kernel<<<8, 256, 0, stream>>>(
            xproj + (size_t)t * 64 * 4096, Whb, hin, hout, cf,
            (float*)d_out, (t == 1023) ? 1 : 0);
        unsigned short* tmp = hin; hin = hout; hout = tmp;
    }
}
